// TrainableLSTMClassifier_48954037240263
// MI455X (gfx1250) — compile-verified, matrix operand order checked by reading, not run on hardware
//
#include <hip/hip_runtime.h>
#include <hip/hip_bf16.h>

// ---------------------------------------------------------------------------
// 2-layer residual LSTM LM forward loss, bf16 WMMA (v_wmma_f32_16x16x32_bf16)
// B=64, T=512 (511 steps), H=1024, VOCAB=256, L=2.
//
// MI455X strategy:
//  * weights fp32->bf16 once into workspace (33.5MB, L2-resident; 192MB L2)
//  * persistent kernel: 64 WGs x 128 threads (256 wave32), one
//    (16-row batch tile x 16-col hidden block x 4 gates) group per wave
//  * activations staged per-WG into LDS via GLOBAL_LOAD_ASYNC_TO_LDS_B128
//    (ASYNCcnt / s_wait_asynccnt), A-fragments via conflict-free ds_load_b128
//    (LDS row stride padded to 1032 elems -> 4-dword bank advance per row)
//  * weight B-fragments stream from L2 as clause'd global_load_b128
//  * grid-wide monotonic-generation barrier sequences the 511 serial steps
// ---------------------------------------------------------------------------

typedef __attribute__((ext_vector_type(16))) __bf16 v16bf;
typedef __attribute__((ext_vector_type(8)))  __bf16 v8bf;
typedef __attribute__((ext_vector_type(8)))  float  v8f;

#define NBLK   64          // workgroups in persistent kernel
#define NTHR   128         // threads per workgroup (4 wave32)
#define Bsz    64
#define Tlen   512
#define STEPS  511
#define Hdim   1024
#define VOCAB  256
#define LDSLD  1032        // padded LDS row stride (elements): conflict-free

#define USE_ASYNC_LDS 1    // gfx1250 async global->LDS copy path

// ---------------- grid-wide barrier (monotonic generation counter) ----------
__device__ __forceinline__ void grid_sync(unsigned* bar) {
    __threadfence();
    __syncthreads();
    if (threadIdx.x == 0) {
        unsigned gen = __hip_atomic_load(&bar[1], __ATOMIC_ACQUIRE, __HIP_MEMORY_SCOPE_AGENT);
        unsigned prev = __hip_atomic_fetch_add(&bar[0], 1u, __ATOMIC_ACQ_REL, __HIP_MEMORY_SCOPE_AGENT);
        if (((prev + 1u) & (NBLK - 1u)) == 0u) {
            __hip_atomic_fetch_add(&bar[1], 1u, __ATOMIC_ACQ_REL, __HIP_MEMORY_SCOPE_AGENT);
        } else {
            while (__hip_atomic_load(&bar[1], __ATOMIC_ACQUIRE, __HIP_MEMORY_SCOPE_AGENT) == gen)
                __builtin_amdgcn_s_sleep(2);
        }
    }
    __syncthreads();
    __threadfence();
}

// ---------------- async stage of one 16x1024 bf16 tile into LDS -------------
// 2048 16-byte chunks spread over the 128 threads of the WG.
__device__ __forceinline__ void stage_tile(__bf16* lds, const __bf16* src) {
    for (int c = threadIdx.x; c < 2048; c += NTHR) {
        int row  = c >> 7;            // 128 chunks per 2048B source row
        int cole = (c & 127) * 8;     // element offset within row
        __bf16*       d = lds + row * LDSLD + cole;
        const __bf16* g = src + (size_t)row * Hdim + cole;
#if USE_ASYNC_LDS
        unsigned laddr = (unsigned)(size_t)d;   // low 32 bits = LDS address
        asm volatile("global_load_async_to_lds_b128 %0, %1, off"
                     :: "v"(laddr), "v"(g) : "memory");
#else
        *(v8bf*)d = *(const v8bf*)g;
#endif
    }
#if USE_ASYNC_LDS
#if __has_builtin(__builtin_amdgcn_s_wait_asynccnt)
    __builtin_amdgcn_s_wait_asynccnt(0);
#else
    asm volatile("s_wait_asynccnt 0x0" ::: "memory");
#endif
#endif
}

// ------------- bf16 fragment loads: two contiguous 16B chunks per lane ------
// 16-bit 16x32 fragment: lane L holds row (L&15); K chunks {hf..hf+7} and
// {16+hf..16+hf+7} where hf = (L>>4)*8.
__device__ __forceinline__ v16bf load_frag_g(const __bf16* base, int ld, int kb) {
    int lane = threadIdx.x & 31;
    int r  = lane & 15;
    int hf = (lane >> 4) << 3;
    const __bf16* p = base + (size_t)r * ld + kb + hf;
    union { v16bf v; v8bf h[2]; } u;
    u.h[0] = *(const v8bf*)(p);
    u.h[1] = *(const v8bf*)(p + 16);
    return u.v;
}
__device__ __forceinline__ v16bf load_frag_lds(const __bf16* lds, int kb) {
    int lane = threadIdx.x & 31;
    int r  = lane & 15;
    int hf = (lane >> 4) << 3;
    const __bf16* p = lds + r * LDSLD + kb + hf;
    union { v16bf v; v8bf h[2]; } u;
    u.h[0] = *(const v8bf*)(p);
    u.h[1] = *(const v8bf*)(p + 16);
    return u.v;
}

__device__ __forceinline__ float sigmoidf_(float x) {
    return 1.0f / (1.0f + __expf(-x));
}
__device__ __forceinline__ float tanh_fast(float x) {
    x = fminf(fmaxf(x, -10.f), 10.f);
    float e = __expf(2.f * x);
    return (e - 1.f) / (e + 1.f);
}

// ---------------- one layer-step for this wave's (rtile, jb) tile group -----
// All 128 threads of the WG enter (rtile uniform per block): LDS staging and
// __syncthreads are block-uniform.
__device__ __forceinline__ void layer_step(
    int l, int rtile, int jb, int lane, __bf16* lds,
    const __bf16* act, const __bf16* hprev, __bf16* hcur,
    const __bf16* resIn, __bf16* resOut,
    const __bf16* wihb, const __bf16* whhb, const float* bias, float* cst)
{
    const int m0 = rtile * 16;
    const int n  = lane & 15;
    const int mh = (lane >> 4) << 3;

    v8f acc[4];
#pragma unroll
    for (int g = 0; g < 4; ++g) {
        float bb = bias[l * 4096 + g * Hdim + jb * 16 + n];
#pragma unroll
        for (int q = 0; q < 8; ++q) acc[g][q] = bb;
    }

    const __bf16* Wl1 = wihb + (size_t)l * 4096 * Hdim;
    const __bf16* Wl2 = whhb + (size_t)l * 4096 * Hdim;

    // ---- gates += x @ Wih[l].T (K=1024), A from LDS ------------------------
    stage_tile(lds, act + (size_t)m0 * Hdim);
    __syncthreads();
    for (int k = 0; k < Hdim; k += 32) {
        v16bf a = load_frag_lds(lds, k);
#pragma unroll
        for (int g = 0; g < 4; ++g) {
            v16bf b = load_frag_g(Wl1 + (size_t)(g * Hdim + jb * 16) * Hdim, Hdim, k);
            acc[g] = __builtin_amdgcn_wmma_f32_16x16x32_bf16(
                false, a, false, b, (short)0, acc[g], false, false);
        }
    }
    __syncthreads();    // all waves done reading before restage

    // ---- gates += h_prev @ Whh[l].T (K=1024), A from LDS -------------------
    stage_tile(lds, hprev + (size_t)m0 * Hdim);
    __syncthreads();
    for (int k = 0; k < Hdim; k += 32) {
        v16bf a = load_frag_lds(lds, k);
#pragma unroll
        for (int g = 0; g < 4; ++g) {
            v16bf b = load_frag_g(Wl2 + (size_t)(g * Hdim + jb * 16) * Hdim, Hdim, k);
            acc[g] = __builtin_amdgcn_wmma_f32_16x16x32_bf16(
                false, a, false, b, (short)0, acc[g], false, false);
        }
    }

    // ---- LSTM cell on C-fragment layout; c state stored fragment-native ----
    float* cp = cst + (((size_t)l * 256 + rtile * 64 + jb) * 32 + lane) * 8;
#pragma unroll
    for (int q = 0; q < 8; ++q) {
        float ig = sigmoidf_(acc[0][q]);
        float fg = sigmoidf_(acc[1][q]);
        float gg = tanh_fast(acc[2][q]);
        float og = sigmoidf_(acc[3][q]);
        float c  = fg * cp[q] + ig * gg;
        cp[q] = c;
        float h = og * tanh_fast(c);
        int m   = m0 + mh + q;
        int col = jb * 16 + n;
        hcur[(size_t)m * Hdim + col] = (__bf16)h;
        float s = h + (float)resIn[(size_t)m * Hdim + col];   // residual
        resOut[(size_t)m * Hdim + col] = (__bf16)s;
    }
}

// ---------------- prep: weight conversion + state/barrier init --------------
__global__ void prep_kernel(const float* Wih, const float* Whh,
                            const float* bih, const float* bhh, const float* Wout,
                            __bf16* wihb, __bf16* whhb, __bf16* woutb, float* bias,
                            __bf16* hbuf, float* cst, unsigned* bar)
{
    size_t i = (size_t)blockIdx.x * blockDim.x + threadIdx.x;
    size_t stride = (size_t)gridDim.x * blockDim.x;
    for (size_t j = i; j < (size_t)2 * 4096 * Hdim; j += stride) {
        wihb[j] = (__bf16)Wih[j];
        whhb[j] = (__bf16)Whh[j];
    }
    for (size_t j = i; j < (size_t)VOCAB * Hdim; j += stride)
        woutb[j] = (__bf16)Wout[j];
    for (size_t j = i; j < 2 * 4096; j += stride)
        bias[j] = bih[j] + bhh[j];
    for (size_t j = i; j < (size_t)4 * Bsz * Hdim; j += stride)
        hbuf[j] = (__bf16)0.f;
    for (size_t j = i; j < (size_t)2 * 256 * 32 * 8; j += stride)
        cst[j] = 0.f;
    if (i == 0) { bar[0] = 0u; bar[1] = 0u; }
}

// ---------------- persistent LSTM kernel ------------------------------------
__global__ void __launch_bounds__(NTHR, 1) lstm_kernel(
    const int* __restrict__ X, const float* __restrict__ embed,
    const float* __restrict__ bout,
    const __bf16* __restrict__ wihb, const __bf16* __restrict__ whhb,
    const __bf16* __restrict__ woutb, const float* __restrict__ bias,
    __bf16* xbuf, __bf16* inp1, __bf16* outb, __bf16* hbuf, float* cst,
    float* logits, float* ce, unsigned* bar)
{
    __shared__ __bf16 ldsA[16 * LDSLD];          // 33 KB staging tile

    const int lane  = threadIdx.x & 31;
    const int wid   = threadIdx.x >> 5;
    const int wg    = blockIdx.x * (NTHR / 32) + wid;   // 0..255
    const int rtile = wg >> 6;                          // batch tile 0..3 (block-uniform)
    const int jb    = wg & 63;                          // hidden block 0..63

    for (int t = 0; t < STEPS; ++t) {
        // ---- phase A: embedding gather x_t = bf16(embed[X[:, t]]) ----------
        {
            int tid  = blockIdx.x * NTHR + threadIdx.x;   // 0..8191
            int row  = tid >> 7;                          // batch row
            int col0 = (tid & 127) * 8;
            int tok  = X[row * Tlen + t];
            const float* e = embed + (size_t)tok * Hdim + col0;
            __bf16* xp = xbuf + (size_t)row * Hdim + col0;
#pragma unroll
            for (int q = 0; q < 8; ++q) xp[q] = (__bf16)e[q];
        }
        grid_sync(bar);

        const __bf16* hp0 = hbuf + (size_t)((t & 1))     * Bsz * Hdim;
        __bf16*       hc0 = hbuf + (size_t)((t & 1) ^ 1) * Bsz * Hdim;
        const __bf16* hp1 = hbuf + (size_t)(2 + (t & 1))       * Bsz * Hdim;
        __bf16*       hc1 = hbuf + (size_t)(2 + ((t & 1) ^ 1)) * Bsz * Hdim;

        // ---- layer 0: inp = x_t, residual out -> inp1 ----------------------
        layer_step(0, rtile, jb, lane, ldsA, xbuf, hp0, hc0, xbuf, inp1,
                   wihb, whhb, bias, cst);
        grid_sync(bar);

        // ---- layer 1: inp = inp1, residual out -> outb ---------------------
        layer_step(1, rtile, jb, lane, ldsA, inp1, hp1, hc1, inp1, outb,
                   wihb, whhb, bias, cst);
        grid_sync(bar);

        // ---- output projection: logits = outb @ Wout.T + bout --------------
        // blocks 0..15 only; rt = wg>>4 is block-uniform -> barriers safe
        if (blockIdx.x < 16) {
            int rt = wg >> 4, vb = wg & 15;
            int n  = lane & 15;
            int mh = (lane >> 4) << 3;
            stage_tile(ldsA, outb + (size_t)(rt * 16) * Hdim);
            __syncthreads();
            v8f acc;
            float bb = bout[vb * 16 + n];
#pragma unroll
            for (int q = 0; q < 8; ++q) acc[q] = bb;
            const __bf16* Bb = woutb + (size_t)(vb * 16) * Hdim;
            for (int k = 0; k < Hdim; k += 32) {
                v16bf a = load_frag_lds(ldsA, k);
                v16bf b = load_frag_g(Bb, Hdim, k);
                acc = __builtin_amdgcn_wmma_f32_16x16x32_bf16(
                    false, a, false, b, (short)0, acc, false, false);
            }
#pragma unroll
            for (int q = 0; q < 8; ++q) {
                int m = rt * 16 + mh + q;
                logits[(size_t)m * VOCAB + vb * 16 + n] = acc[q];
            }
        }
        grid_sync(bar);

        // ---- CE: log-softmax + gather target (64 rows on WG0) --------------
        if (blockIdx.x == 0 && threadIdx.x < Bsz) {
            int b = threadIdx.x;
            const float* lg = logits + (size_t)b * VOCAB;
            float mx = -1e30f;
            for (int j = 0; j < VOCAB; ++j) mx = fmaxf(mx, lg[j]);
            float s = 0.f;
            for (int j = 0; j < VOCAB; ++j) s += __expf(lg[j] - mx);
            float lse = mx + __logf(s);
            int tgt = X[b * Tlen + t + 1];
            ce[(size_t)t * Bsz + b] = lse - lg[tgt];
        }
        grid_sync(bar);
    }
}

// ---------------- final reduction: loss = sum(ce) / (B * T) -----------------
__global__ void reduce_kernel(const float* __restrict__ ce, float* out)
{
    __shared__ float sh[256];
    float s = 0.f;
    for (int i = threadIdx.x; i < STEPS * Bsz; i += 256) s += ce[i];
    sh[threadIdx.x] = s;
    __syncthreads();
    for (int w = 128; w > 0; w >>= 1) {
        if ((int)threadIdx.x < w) sh[threadIdx.x] += sh[threadIdx.x + w];
        __syncthreads();
    }
    if (threadIdx.x == 0) out[0] = sh[0] / ((float)Bsz * (float)Tlen);
}

// ---------------------------------------------------------------------------
extern "C" void kernel_launch(void* const* d_in, const int* in_sizes, int n_in,
                              void* d_out, int out_size, void* d_ws, size_t ws_size,
                              hipStream_t stream)
{
    const int*   X     = (const int*)  d_in[0];
    const float* embed = (const float*)d_in[1];
    const float* Wih   = (const float*)d_in[2];
    const float* Whh   = (const float*)d_in[3];
    const float* bih   = (const float*)d_in[4];
    const float* bhh   = (const float*)d_in[5];
    const float* Wout  = (const float*)d_in[6];
    const float* bout  = (const float*)d_in[7];

    char* ws = (char*)d_ws;
    size_t off = 0;
    auto alloc = [&](size_t bytes) -> char* {
        char* p = ws + off;
        off += (bytes + 255) & ~(size_t)255;
        return p;
    };

    unsigned* bar   = (unsigned*)alloc(256);
    __bf16*   wihb  = (__bf16*)  alloc((size_t)2 * 4096 * Hdim * 2);
    __bf16*   whhb  = (__bf16*)  alloc((size_t)2 * 4096 * Hdim * 2);
    __bf16*   woutb = (__bf16*)  alloc((size_t)VOCAB * Hdim * 2);
    float*    bias  = (float*)   alloc((size_t)2 * 4096 * 4);
    __bf16*   xbuf  = (__bf16*)  alloc((size_t)Bsz * Hdim * 2);
    __bf16*   inp1  = (__bf16*)  alloc((size_t)Bsz * Hdim * 2);
    __bf16*   outb  = (__bf16*)  alloc((size_t)Bsz * Hdim * 2);
    __bf16*   hbuf  = (__bf16*)  alloc((size_t)4 * Bsz * Hdim * 2);
    float*    cst   = (float*)   alloc((size_t)2 * 256 * 32 * 8 * 4);
    float*    logits= (float*)   alloc((size_t)Bsz * VOCAB * 4);
    float*    ce    = (float*)   alloc((size_t)STEPS * Bsz * 4);
    (void)ws_size; (void)in_sizes; (void)n_in; (void)out_size;

    prep_kernel<<<1024, 256, 0, stream>>>(Wih, Whh, bih, bhh, Wout,
                                          wihb, whhb, woutb, bias,
                                          hbuf, cst, bar);
    lstm_kernel<<<NBLK, NTHR, 0, stream>>>(X, embed, bout,
                                           wihb, whhb, woutb, bias,
                                           xbuf, inp1, outb, hbuf, cst,
                                           logits, ce, bar);
    reduce_kernel<<<1, 256, 0, stream>>>(ce, (float*)d_out);
}
